// ProsodyEncoder_72782515798283
// MI455X (gfx1250) — matrix operand order repaired and checked by reading, not compiled, not run on hardware
//
#include <hip/hip_runtime.h>
#include <hip/hip_bf16.h>

typedef __attribute__((ext_vector_type(16))) __bf16 v16bf;
typedef __attribute__((ext_vector_type(8)))  float  v8f;

#define NUM_EMB 128
#define EMB_DIM 256
#define ROWS_PER_BLOCK 128
#define N_TOTAL (64 * 2048) /* 131072 */
#define NBLOCKS (N_TOTAL / ROWS_PER_BLOCK) /* 1024 */
#define B_ELEMS (NUM_EMB * EMB_DIM) /* 32768 bf16 per plane */

// ---------------------------------------------------------------------------
// Init: zero global histogram + precompute ||e_k||^2 (ws is poisoned once and
// never re-poisoned between graph replays, so re-init every launch).
// ---------------------------------------------------------------------------
__global__ void vq_init(int* __restrict__ counts,
                        const float* __restrict__ emb,
                        float* __restrict__ esq_ws) {
    const int t = threadIdx.x;
    if (t < NUM_EMB) {
        counts[t] = 0;
        const float4* ep = (const float4*)(emb + t * EMB_DIM);
        float s = 0.f;
        #pragma unroll
        for (int i = 0; i < EMB_DIM / 4; ++i) {
            float4 v = ep[i];
            s += v.x * v.x + v.y * v.y + v.z * v.z + v.w * v.w;
        }
        esq_ws[t] = s;
    }
}

// ---------------------------------------------------------------------------
// Prep: convert codebook f32 -> bf16 hi/lo ONCE, stored in WMMA B-operand
// layout. Tile t = c*8+s holds 32 lanes x 16 elems:
//   value(t, lane, e) = emb[(t>>3)*16 + (lane&15)][(t&7)*32 + (lane>>4)*16 + e]
// Main kernel then fetches B tiles with plain 32B vector loads (L2-resident).
// ---------------------------------------------------------------------------
__global__ void vq_prep(const float* __restrict__ emb,
                        __bf16* __restrict__ bh, __bf16* __restrict__ bl) {
    const int id = blockIdx.x * 256 + threadIdx.x;   // 32768 total
    const int t    = id >> 9;
    const int lane = (id >> 4) & 31;
    const int e    = id & 15;
    const int code = (t >> 3) * 16 + (lane & 15);
    const int k    = (t & 7) * 32 + (lane >> 4) * 16 + e;
    const float v  = emb[code * EMB_DIM + k];
    const __bf16 h = (__bf16)v;
    bh[id] = h;
    bl[id] = (__bf16)(v - (float)h);
}

// ---------------------------------------------------------------------------
// Main: 8 waves x 16-row tiles; compensated bf16 WMMA scoring against all 128
// codes, cross-lane argmin, then vectorized emission of quantized / one-hot /
// indices / loss partial / histogram.
// ---------------------------------------------------------------------------
__global__ __launch_bounds__(256) void vq_main(
    const float*  __restrict__ inp,          // [N, 256]
    const float*  __restrict__ emb,          // [128, 256]
    const float*  __restrict__ esq_ws,       // [128]
    const __bf16* __restrict__ bh,           // [64 tiles][512] WMMA-B layout
    const __bf16* __restrict__ bl,
    float* __restrict__ loss_partials,       // [NBLOCKS]
    int*   __restrict__ counts,              // [128]
    float* __restrict__ qout,                // d_out+1 (4B-aligned only)
    float* __restrict__ encout,              // d_out+2+Q (8B-aligned, k%4==2)
    float* __restrict__ idxout)              // [N] as float
{
    __shared__ float esq[NUM_EMB];
    __shared__ int   hist[NUM_EMB];
    __shared__ int   idx_l[ROWS_PER_BLOCK];
    __shared__ float red[256];

    const int  tid  = threadIdx.x;
    const int  lane = tid & 31;
    const int  wave = tid >> 5;
    const long blk  = blockIdx.x;

    if (tid < NUM_EMB) {
        esq[tid]  = esq_ws[tid];
        hist[tid] = 0;
    }
    __syncthreads();

    // ---- build A (16 rows x 256) bf16 hi/lo in WMMA A layout ----
    const int  mrow    = lane & 15;
    const int  hiHalf  = lane >> 4;
    const long rowBase = blk * ROWS_PER_BLOCK + (long)wave * 16;

    v16bf a_hi[8], a_lo[8];
    {
        const float* arow = inp + (rowBase + mrow) * EMB_DIM;
        #pragma unroll
        for (int s = 0; s < 8; ++s) {
            const int kb = s * 32 + hiHalf * 16;
            float4 v0a = *(const float4*)(arow + kb);
            float4 v0b = *(const float4*)(arow + kb + 4);
            float4 v1a = *(const float4*)(arow + kb + 8);
            float4 v1b = *(const float4*)(arow + kb + 12);
            float f16v[16] = {v0a.x, v0a.y, v0a.z, v0a.w, v0b.x, v0b.y, v0b.z, v0b.w,
                              v1a.x, v1a.y, v1a.z, v1a.w, v1b.x, v1b.y, v1b.z, v1b.w};
            #pragma unroll
            for (int e = 0; e < 16; ++e) {
                __bf16 h = (__bf16)f16v[e];
                a_hi[s][e] = h;
                a_lo[s][e] = (__bf16)(f16v[e] - (float)h);
            }
        }
    }

    // ---- score 8 column tiles; per-lane running argmin ----
    float minv[8];
    int   mini[8];
    #pragma unroll
    for (int j = 0; j < 8; ++j) { minv[j] = 3.4e38f; mini[j] = 0; }

    #pragma unroll 1
    for (int c = 0; c < 8; ++c) {
        v8f acc = {};
        #pragma unroll
        for (int s = 0; s < 8; ++s) {
            const int off = (c * 8 + s) * 512 + lane * 16;  // 32B-aligned
            v16bf b_hi = *(const v16bf*)(bh + off);
            v16bf b_lo = *(const v16bf*)(bl + off);
            acc = __builtin_amdgcn_wmma_f32_16x16x32_bf16(false, a_hi[s], false, b_hi,
                                                          (short)0, acc, false, false);
            acc = __builtin_amdgcn_wmma_f32_16x16x32_bf16(false, a_lo[s], false, b_hi,
                                                          (short)0, acc, false, false);
            acc = __builtin_amdgcn_wmma_f32_16x16x32_bf16(false, a_hi[s], false, b_lo,
                                                          (short)0, acc, false, false);
        }
        const int   code = c * 16 + mrow;       // C layout: lane = code column
        const float eq   = esq[code];
        #pragma unroll
        for (int j = 0; j < 8; ++j) {
            float d = eq - 2.0f * acc[j];       // ||e||^2 - 2 x.e (||x||^2 const)
            if (d < minv[j]) { minv[j] = d; mini[j] = code; }
        }
    }

    // ---- cross-lane argmin within each 16-lane group ----
    #pragma unroll
    for (int off = 8; off >= 1; off >>= 1) {
        #pragma unroll
        for (int j = 0; j < 8; ++j) {
            float ov = __shfl_xor(minv[j], off, 16);
            int   oi = __shfl_xor(mini[j], off, 16);
            if (ov < minv[j] || (ov == minv[j] && oi < mini[j])) {
                minv[j] = ov; mini[j] = oi;
            }
        }
    }

    if (mrow == 0) {  // lanes 0 (rows 0-7) / 16 (rows 8-15)
        #pragma unroll
        for (int j = 0; j < 8; ++j) {
            const int r = wave * 16 + j + hiHalf * 8;
            idx_l[r] = mini[j];
            atomicAdd(&hist[mini[j]], 1);
            idxout[blk * ROWS_PER_BLOCK + r] = (float)mini[j];
        }
    }
    __syncthreads();

    // ---- quantized (offset+1 -> rolling-window b128 stores) + loss ----
    const int  row  = tid >> 1;
    const int  half = tid & 1;
    const int  idx  = idx_l[row];
    const long R    = blk * ROWS_PER_BLOCK + row;
    float ls = 0.f;
    {
        const float4* ep4   = (const float4*)(emb + (long)idx * EMB_DIM);
        const float4* xp4   = (const float4*)(inp + R * EMB_DIM);
        float*        qbase = qout + R * EMB_DIM;   // col k at qbase[k]; aligned iff k%4==3

        if (half == 0) {
            // cols 0..127 for loss; stores: scalars 0,1,2 + b128 at k=3+4i (i=0..31)
            float4 cur = ep4[0];
            { float4 x = xp4[0];
              float dx=cur.x-x.x, dy=cur.y-x.y, dz=cur.z-x.z, dw=cur.w-x.w;
              ls += dx*dx+dy*dy+dz*dz+dw*dw; }
            qbase[0] = cur.x; qbase[1] = cur.y; qbase[2] = cur.z;
            #pragma unroll
            for (int i = 0; i < 32; ++i) {
                float4 nxt = ep4[i + 1];        // i==31 reads into half1 region (ok)
                if (i < 31) {
                    float4 x = xp4[i + 1];
                    float dx=nxt.x-x.x, dy=nxt.y-x.y, dz=nxt.z-x.z, dw=nxt.w-x.w;
                    ls += dx*dx+dy*dy+dz*dz+dw*dw;
                }
                *(float4*)(qbase + 4 * i + 3) = make_float4(cur.w, nxt.x, nxt.y, nxt.z);
                cur = nxt;
            }
        } else {
            // cols 128..255; stores: b128 at k=4*(31+i)+3 (i=1..31) + scalar 255
            float4 cur = ep4[32];
            { float4 x = xp4[32];
              float dx=cur.x-x.x, dy=cur.y-x.y, dz=cur.z-x.z, dw=cur.w-x.w;
              ls += dx*dx+dy*dy+dz*dz+dw*dw; }
            #pragma unroll
            for (int i = 1; i < 32; ++i) {
                float4 nxt = ep4[32 + i];
                float4 x   = xp4[32 + i];
                float dx=nxt.x-x.x, dy=nxt.y-x.y, dz=nxt.z-x.z, dw=nxt.w-x.w;
                ls += dx*dx+dy*dy+dz*dz+dw*dw;
                *(float4*)(qbase + 4 * (32 + i) - 1) = make_float4(cur.w, nxt.x, nxt.y, nxt.z);
                cur = nxt;
            }
            qbase[255] = cur.w;
        }
    }

    // ---- one-hot encodings (offset%4==2 -> b128 at k=2+4i) ----
    {
        float* encb = encout + R * NUM_EMB;     // col k aligned iff k%4==2
        if (half == 0) {
            encb[0] = (float)(idx == 0);
            encb[1] = (float)(idx == 1);
            #pragma unroll
            for (int i = 0; i < 16; ++i) {
                const int k = 2 + 4 * i;
                *(float4*)(encb + k) = make_float4((float)(idx == k),
                                                   (float)(idx == k + 1),
                                                   (float)(idx == k + 2),
                                                   (float)(idx == k + 3));
            }
        } else {
            #pragma unroll
            for (int i = 16; i < 31; ++i) {
                const int k = 2 + 4 * i;
                *(float4*)(encb + k) = make_float4((float)(idx == k),
                                                   (float)(idx == k + 1),
                                                   (float)(idx == k + 2),
                                                   (float)(idx == k + 3));
            }
            encb[126] = (float)(idx == 126);
            encb[127] = (float)(idx == 127);
        }
    }

    red[tid] = ls;
    __syncthreads();
    for (int off = 128; off >= 1; off >>= 1) {
        if (tid < off) red[tid] += red[tid + off];
        __syncthreads();
    }
    if (tid == 0) loss_partials[blk] = red[0];
    if (tid < NUM_EMB) {
        int h = hist[tid];
        if (h) atomicAdd(&counts[tid], h);
    }
}

// ---------------------------------------------------------------------------
// Final: fixed-order loss sum + perplexity from integer counts.
// ---------------------------------------------------------------------------
__global__ __launch_bounds__(256) void vq_final(
    const float* __restrict__ loss_partials, int nblocks,
    const int*   __restrict__ counts,
    float* __restrict__ loss_out, float* __restrict__ perp_out)
{
    __shared__ float red[256];
    const int tid = threadIdx.x;

    float s = 0.f;
    for (int i = tid; i < nblocks; i += 256) s += loss_partials[i];
    red[tid] = s;
    __syncthreads();
    for (int off = 128; off >= 1; off >>= 1) {
        if (tid < off) red[tid] += red[tid + off];
        __syncthreads();
    }
    const float lossSum = red[0];
    __syncthreads();

    float t = 0.f;
    if (tid < NUM_EMB) {
        float avg = (float)counts[tid] / (float)N_TOTAL;
        t = avg * logf(avg + 1e-10f);
    }
    red[tid] = t;
    __syncthreads();
    for (int off = 128; off >= 1; off >>= 1) {
        if (tid < off) red[tid] += red[tid + off];
        __syncthreads();
    }
    if (tid == 0) {
        *loss_out = 0.25f * lossSum / (float)((long)N_TOTAL * EMB_DIM);
        *perp_out = expf(-red[0]);
    }
}

extern "C" void kernel_launch(void* const* d_in, const int* in_sizes, int n_in,
                              void* d_out, int out_size, void* d_ws, size_t ws_size,
                              hipStream_t stream) {
    const float* inp = (const float*)d_in[0];  // [64,2048,256] f32
    const float* emb = (const float*)d_in[1];  // [128,256]     f32
    float* out = (float*)d_out;

    // Output tuple layout (flat, return order):
    //   [0] loss | [1..1+Q) quantized_st | [1+Q] perplexity |
    //   [2+Q..2+Q+N*128) encodings | then indices (as float)
    const long Q = (long)N_TOTAL * EMB_DIM;
    float* loss_out = out;
    float* qout     = out + 1;
    float* perp_out = out + 1 + Q;
    float* encout   = out + 2 + Q;
    float* idxout   = out + 2 + Q + (long)N_TOTAL * NUM_EMB;

    // Workspace layout
    float*  loss_partials = (float*)d_ws;                    // [1024]
    int*    counts        = (int*)(loss_partials + NBLOCKS); // [128]
    float*  esq_ws        = (float*)(counts + NUM_EMB);      // [128]
    __bf16* bh            = (__bf16*)(esq_ws + NUM_EMB);     // [32768] (32B-aligned)
    __bf16* bl            = bh + B_ELEMS;                    // [32768]

    vq_init<<<1, 128, 0, stream>>>(counts, emb, esq_ws);
    vq_prep<<<B_ELEMS / 256, 256, 0, stream>>>(emb, bh, bl);
    vq_main<<<NBLOCKS, 256, 0, stream>>>(inp, emb, esq_ws, bh, bl,
                                         loss_partials, counts,
                                         qout, encout, idxout);
    vq_final<<<1, 256, 0, stream>>>(loss_partials, NBLOCKS, counts,
                                    loss_out, perp_out);
}
